// Yolo_v1_1_2542620639989
// MI455X (gfx1250) — matrix operand order repaired and checked by reading, not compiled, not run on hardware
//
#include <hip/hip_runtime.h>
#include <cstdint>

// ---------------------------------------------------------------------------
// Types for WMMA fragments (gfx1250 / CDNA5, wave32)
// ---------------------------------------------------------------------------
typedef __attribute__((ext_vector_type(16))) __bf16       v16bf;
typedef __attribute__((ext_vector_type(8)))  float        v8f;
typedef __attribute__((ext_vector_type(4)))  unsigned int u32x4;

union Frag {
    v16bf v;
    u32x4 u[2];
};

__device__ __forceinline__ unsigned short f2bf(float f) {
    union { float f; unsigned int u; } a{f};
    return (unsigned short)((a.u + 0x7FFFu + ((a.u >> 16) & 1u)) >> 16);
}
__device__ __forceinline__ unsigned int f2bf_pair(float lo, float hi) {
    union { float f; unsigned int u; } a{lo}, b{hi};
    unsigned int ra = (a.u + 0x7FFFu + ((a.u >> 16) & 1u)) >> 16;
    unsigned int rb = (b.u + 0x7FFFu + ((b.u >> 16) & 1u)) >> 16;
    return (ra & 0xFFFFu) | (rb << 16);
}

// ---------------------------------------------------------------------------
// Zero-fill (for halo-padded activation buffers; 16B granules)
// ---------------------------------------------------------------------------
__global__ void __launch_bounds__(256)
zero_u32x4(u32x4* __restrict__ p, long long n16) {
    long long i = (long long)blockIdx.x * 256 + threadIdx.x;
    if (i < n16) p[i] = (u32x4)0u;
}

// ---------------------------------------------------------------------------
// NCHW fp32 -> halo-padded NHWC bf16 interior  [B][H+2][W+2][C]
// ---------------------------------------------------------------------------
__global__ void __launch_bounds__(256)
nchw_to_nhwc_pad_bf16(const float* __restrict__ in, unsigned short* __restrict__ out,
                      int B, int C, int H, int W) {
    long long idx = (long long)blockIdx.x * 256 + threadIdx.x;
    long long total = (long long)B * C * H * W;
    if (idx >= total) return;
    int c = (int)(idx % C);
    long long t = idx / C;
    int w = (int)(t % W);
    t /= W;
    int h = (int)(t % H);
    int b = (int)(t / H);
    out[(((long long)b * (H + 2) + h + 1) * (W + 2) + w + 1) * C + c] =
        f2bf(in[(((long long)b * C + c) * H + h) * W + w]);
}

// ---------------------------------------------------------------------------
// Fold BN scale into weights; repack [O][C][3][3] fp32 -> [kk][Opad][C] bf16
// (o >= O rows zero-padded for the 50->64 head)
// ---------------------------------------------------------------------------
__global__ void __launch_bounds__(256)
pack_weights_bf16(const float* __restrict__ w, const float* __restrict__ g,
                  const float* __restrict__ v, int O, int Opad, int C,
                  unsigned short* __restrict__ wp) {
    long long idx = (long long)blockIdx.x * 256 + threadIdx.x;
    long long total = (long long)9 * Opad * C;
    if (idx >= total) return;
    int c = (int)(idx % C);
    long long t = idx / C;
    int o = (int)(t % Opad);
    int kk = (int)(t / Opad);
    float val = 0.0f;
    if (o < O) {
        float scale = g[o] * rsqrtf(v[o] + 1e-5f);
        val = w[((long long)o * C + c) * 9 + kk] * scale;
    }
    wp[((long long)kk * Opad + o) * C + c] = f2bf(val);
}

__global__ void __launch_bounds__(256)
make_bias(const float* __restrict__ b, const float* __restrict__ g,
          const float* __restrict__ be, const float* __restrict__ mu,
          const float* __restrict__ v, int O, int Opad, float* __restrict__ bias) {
    int o = blockIdx.x * 256 + threadIdx.x;
    if (o >= Opad) return;
    float r = 0.0f;
    if (o < O) {
        float scale = g[o] * rsqrtf(v[o] + 1e-5f);
        r = (b[o] - mu[o]) * scale + be[o];
    }
    bias[o] = r;
}

// ---------------------------------------------------------------------------
// Implicit-GEMM 3x3 conv on halo-padded NHWC bf16 with v_wmma_f32_16x16x32_bf16.
// One wave32 per 32(M=out-ch) x 64(N=b*ho*wo) tile; all dims compile-time so the
// steady state is pure {global_load_b128 imm-offset} + {v_wmma}: no address
// VALU, no validity predicates (halo reads hit real zeros). Base pointers are
// kept as SCALAR variables (not arrays) so addrspace inference yields
// global_load (not flat_load) for every stream.
// Lane layouts used by this kernel (wave32; A and B share this K order):
//   A 16x32: lane(lm,hf) row m=lm, K = hf*16+{0..15}
//   B 32x16: lane col n=lm, K = hf*16+{0..15}
//   C/D:     lane col n=lm, VGPR j -> row m = hf*8 + j
// ---------------------------------------------------------------------------
template <int STRIDE, bool FINAL, int Cin, int Hin, int Win, int Opad, int Hout, int Wout>
__global__ void __launch_bounds__(32)
conv3x3_bn_act_wmma(const unsigned short* __restrict__ in,   // [B][Hin+2][Win+2][Cin] bf16
                    const unsigned short* __restrict__ wp,   // [9][Opad][Cin] bf16
                    const float* __restrict__ bias,          // [Opad]
                    unsigned short* __restrict__ out_bf,     // [B][Hout+2][Wout+2][Opad] bf16
                    float* __restrict__ out_f32) {           // FINAL: [Npos][50] fp32
    constexpr int Wpad = Win + 2;
    constexpr int Hpad = Hin + 2;
    const int lane  = threadIdx.x;
    const int lm    = lane & 15;
    const int hf    = lane >> 4;
    const int obase = blockIdx.x * 32;
    const int nbase = blockIdx.y * 64;

    // per-lane base offset at (kh=0, kw=0) in the padded input
    auto inbase = [&](int ni) -> const unsigned short* {
        int n  = nbase + ni * 16 + lm;
        int wo = n % Wout;
        int t  = n / Wout;
        int ho = t % Hout;
        int b  = t / Hout;
        return in + (((long long)b * Hpad + ho * STRIDE) * Wpad + wo * STRIDE) * Cin
                  + hf * 16;
    };
    const unsigned short* pin0 = inbase(0);
    const unsigned short* pin1 = inbase(1);
    const unsigned short* pin2 = inbase(2);
    const unsigned short* pin3 = inbase(3);

    const unsigned short* pw0 = wp + (long long)(obase + lm) * Cin + hf * 16;
    const unsigned short* pw1 = wp + (long long)(obase + 16 + lm) * Cin + hf * 16;

    v8f acc[2][4];
#pragma unroll
    for (int mi = 0; mi < 2; ++mi)
#pragma unroll
        for (int ni = 0; ni < 4; ++ni)
            acc[mi][ni] = (v8f)0.0f;

#pragma unroll 1
    for (int kh = 0; kh < 3; ++kh) {
#pragma unroll 1
        for (int kw = 0; kw < 3; ++kw) {
#pragma unroll
            for (int cb = 0; cb < Cin; cb += 32) {
                Frag a0, a1;
                a0.u[0] = *(const u32x4*)(pw0 + cb);
                a0.u[1] = *(const u32x4*)(pw0 + cb + 8);
                a1.u[0] = *(const u32x4*)(pw1 + cb);
                a1.u[1] = *(const u32x4*)(pw1 + cb + 8);
                Frag b0, b1, b2, b3;
                b0.u[0] = *(const u32x4*)(pin0 + cb);
                b0.u[1] = *(const u32x4*)(pin0 + cb + 8);
                b1.u[0] = *(const u32x4*)(pin1 + cb);
                b1.u[1] = *(const u32x4*)(pin1 + cb + 8);
                b2.u[0] = *(const u32x4*)(pin2 + cb);
                b2.u[1] = *(const u32x4*)(pin2 + cb + 8);
                b3.u[0] = *(const u32x4*)(pin3 + cb);
                b3.u[1] = *(const u32x4*)(pin3 + cb + 8);

                acc[0][0] = __builtin_amdgcn_wmma_f32_16x16x32_bf16(
                    false, a0.v, false, b0.v, (short)0, acc[0][0], false, false);
                acc[1][0] = __builtin_amdgcn_wmma_f32_16x16x32_bf16(
                    false, a1.v, false, b0.v, (short)0, acc[1][0], false, false);
                acc[0][1] = __builtin_amdgcn_wmma_f32_16x16x32_bf16(
                    false, a0.v, false, b1.v, (short)0, acc[0][1], false, false);
                acc[1][1] = __builtin_amdgcn_wmma_f32_16x16x32_bf16(
                    false, a1.v, false, b1.v, (short)0, acc[1][1], false, false);
                acc[0][2] = __builtin_amdgcn_wmma_f32_16x16x32_bf16(
                    false, a0.v, false, b2.v, (short)0, acc[0][2], false, false);
                acc[1][2] = __builtin_amdgcn_wmma_f32_16x16x32_bf16(
                    false, a1.v, false, b2.v, (short)0, acc[1][2], false, false);
                acc[0][3] = __builtin_amdgcn_wmma_f32_16x16x32_bf16(
                    false, a0.v, false, b3.v, (short)0, acc[0][3], false, false);
                acc[1][3] = __builtin_amdgcn_wmma_f32_16x16x32_bf16(
                    false, a1.v, false, b3.v, (short)0, acc[1][3], false, false);
            }
            // advance to next tap
            pw0 += Opad * Cin;
            pw1 += Opad * Cin;
            pin0 += Cin; pin1 += Cin; pin2 += Cin; pin3 += Cin;
        }
        pin0 += (Wpad - 3) * Cin;
        pin1 += (Wpad - 3) * Cin;
        pin2 += (Wpad - 3) * Cin;
        pin3 += (Wpad - 3) * Cin;
    }

    // ---- epilogue: bias + activation ----
    float bv[2][8];
#pragma unroll
    for (int mi = 0; mi < 2; ++mi)
#pragma unroll
        for (int j = 0; j < 8; ++j)
            bv[mi][j] = bias[obase + mi * 16 + hf * 8 + j];

#pragma unroll
    for (int ni = 0; ni < 4; ++ni) {
        const int n  = nbase + ni * 16 + lm;
        const int wo = n % Wout;
        const int t  = n / Wout;
        const int ho = t % Hout;
        const int b  = t / Hout;
#pragma unroll
        for (int mi = 0; mi < 2; ++mi) {
            if (!FINAL) {
                float tv[8];
#pragma unroll
                for (int j = 0; j < 8; ++j) {
                    float x = acc[mi][ni][j] + bv[mi][j];
                    tv[j] = (x >= 0.0f) ? x : 0.01f * x;   // LeakyReLU(0.01)
                }
                u32x4 st;
                st.x = f2bf_pair(tv[0], tv[1]);
                st.y = f2bf_pair(tv[2], tv[3]);
                st.z = f2bf_pair(tv[4], tv[5]);
                st.w = f2bf_pair(tv[6], tv[7]);
                unsigned short* po = out_bf +
                    (((long long)b * (Hout + 2) + ho + 1) * (Wout + 2) + wo + 1) * Opad +
                    obase + mi * 16 + hf * 8;
                *(u32x4*)po = st;
            } else {
#pragma unroll
                for (int j = 0; j < 8; ++j) {
                    int o = obase + mi * 16 + hf * 8 + j;
                    if (o < 50) {
                        float x = acc[mi][ni][j] + bv[mi][j];
                        x = 1.0f / (1.0f + expf(-x));       // sigmoid
                        out_f32[(long long)n * 50 + o] = x;
                    }
                }
            }
        }
    }
}

// ---------------------------------------------------------------------------
// Launch sequence (single stream, graph-capture safe):
//   zero halo buffer -> write interior -> conv -> ... ; final layer writes
//   fp32 NHWC [B,7,7,50] straight into d_out (== reference permute).
// ---------------------------------------------------------------------------
extern "C" void kernel_launch(void* const* d_in, const int* in_sizes, int n_in,
                              void* d_out, int out_size, void* d_ws, size_t ws_size,
                              hipStream_t stream) {
    (void)in_sizes; (void)n_in; (void)out_size; (void)ws_size;
    const int B = 64, DIM_IN = 512, DL = 1024;

    char* ws = (char*)d_ws;
    const size_t ACT_BYTES = 34ull * 1024 * 1024;  // >= 64*16*16*1024*2 = 32MiB
    const size_t WP_BYTES  = 20ull * 1024 * 1024;  // >= 9*1024*1024*2  = 18MiB
    unsigned short* actA   = (unsigned short*)(ws);
    unsigned short* actB   = (unsigned short*)(ws + ACT_BYTES);
    unsigned short* wpack  = (unsigned short*)(ws + 2 * ACT_BYTES);
    float*          biasbf = (float*)(ws + 2 * ACT_BYTES + WP_BYTES);

    const float* x = (const float*)d_in[0];
    auto W_ = [&](int li) { return (const float*)d_in[1 + 6 * li + 0]; };
    auto Bc = [&](int li) { return (const float*)d_in[1 + 6 * li + 1]; };
    auto G_ = [&](int li) { return (const float*)d_in[1 + 6 * li + 2]; };
    auto Be = [&](int li) { return (const float*)d_in[1 + 6 * li + 3]; };
    auto Mu = [&](int li) { return (const float*)d_in[1 + 6 * li + 4]; };
    auto V_ = [&](int li) { return (const float*)d_in[1 + 6 * li + 5]; };

    auto zero = [&](void* p, long long bytes) {
        long long n16 = bytes / 16;
        zero_u32x4<<<(int)((n16 + 255) / 256), 256, 0, stream>>>((u32x4*)p, n16);
    };
    auto pack = [&](int li, int O, int Opad, int C) {
        long long tw = (long long)9 * Opad * C;
        pack_weights_bf16<<<(int)((tw + 255) / 256), 256, 0, stream>>>(
            W_(li), G_(li), V_(li), O, Opad, C, wpack);
        make_bias<<<(Opad + 255) / 256, 256, 0, stream>>>(
            Bc(li), G_(li), Be(li), Mu(li), V_(li), O, Opad, biasbf);
    };

    const long long PAD512_14  = 64LL * 16 * 16 * 512 * 2;   // 16 MiB
    const long long PAD1024_16 = 64LL * 16 * 16 * 1024 * 2;  // 32 MiB
    const long long PAD1024_9  = 64LL * 9 * 9 * 1024 * 2;    // ~10 MiB

    const int N14 = B * 14 * 14;  // 12544 = 196*64
    const int N7  = B * 7 * 7;    // 3136  = 49*64

    // input: NCHW fp32 -> padded NHWC bf16 (actA)
    zero(actA, PAD512_14);
    {
        long long total = (long long)B * DIM_IN * 14 * 14;
        nchw_to_nhwc_pad_bf16<<<(int)((total + 255) / 256), 256, 0, stream>>>(
            x, actA, B, DIM_IN, 14, 14);
    }

    // L0: 512 -> 1024, 14x14, stride 1   (actA -> actB)
    pack(0, DL, DL, DIM_IN);
    zero(actB, PAD1024_16);
    conv3x3_bn_act_wmma<1, false, 512, 14, 14, 1024, 14, 14>
        <<<dim3(DL / 32, N14 / 64), 32, 0, stream>>>(actA, wpack, biasbf, actB, nullptr);

    // L1: 1024 -> 1024, stride 2, 14x14 -> 7x7   (actB -> actA)
    pack(1, DL, DL, DL);
    zero(actA, PAD1024_9);
    conv3x3_bn_act_wmma<2, false, 1024, 14, 14, 1024, 7, 7>
        <<<dim3(DL / 32, N7 / 64), 32, 0, stream>>>(actB, wpack, biasbf, actA, nullptr);

    // L2: 1024 -> 1024, 7x7   (actA -> actB)
    pack(2, DL, DL, DL);
    zero(actB, PAD1024_9);
    conv3x3_bn_act_wmma<1, false, 1024, 7, 7, 1024, 7, 7>
        <<<dim3(DL / 32, N7 / 64), 32, 0, stream>>>(actA, wpack, biasbf, actB, nullptr);

    // L3: 1024 -> 1024, 7x7   (actB -> actA)
    pack(3, DL, DL, DL);
    zero(actA, PAD1024_9);
    conv3x3_bn_act_wmma<1, false, 1024, 7, 7, 1024, 7, 7>
        <<<dim3(DL / 32, N7 / 64), 32, 0, stream>>>(actB, wpack, biasbf, actA, nullptr);

    // L4 (head): 1024 -> 50 (padded to 64), sigmoid, fp32 NHWC -> d_out
    pack(4, 50, 64, DL);
    conv3x3_bn_act_wmma<1, true, 1024, 7, 7, 64, 7, 7>
        <<<dim3(64 / 32, N7 / 64), 32, 0, stream>>>(actA, wpack, biasbf, nullptr,
                                                    (float*)d_out);
}
